// VisionMambaViM_17111149707737
// MI455X (gfx1250) — matrix operand order repaired and checked by reading, not compiled, not run on hardware
//
#include <hip/hip_runtime.h>
#include <hip/hip_bf16.h>
#include <cstddef>
#include <cstdint>

// ---------------------------------------------------------------------------
// Types for CDNA5 WMMA (wave32): v_wmma_f32_16x16x32_bf16
// ---------------------------------------------------------------------------
typedef float   v8f     __attribute__((ext_vector_type(8)));
typedef __bf16  bf16x16 __attribute__((ext_vector_type(16)));
typedef __bf16  bf16x8  __attribute__((ext_vector_type(8)));
typedef __bf16  bf16x4  __attribute__((ext_vector_type(4)));

#define DEV_INLINE __device__ __forceinline__

DEV_INLINE float dev_silu(float x)     { return x / (1.0f + __expf(-x)); }
DEV_INLINE float dev_softplus(float x) { return (x > 20.0f) ? x : log1pf(__expf(x)); }

// Async global->LDS copy (CDNA5 GLOBAL_LOAD_ASYNC_TO_LDS_B128, ASYNCcnt) via
// inline asm — portable across toolchains, bypasses builtin signature drift.
#if defined(__gfx1250__)
#define HAVE_ASYNC_LDS 1
#else
#define HAVE_ASYNC_LDS 0
#endif

// ---------------------------------------------------------------------------
// FAST WMMA GEMM:  C[M,N] = A[M,K] * W[N,K]^T   (+bias, +accumulate, +epi)
// Preconditions (checked by host): M%32==0, N%64==0, K%64==0, lda%8==0, ldw%8==0.
// Block: 256 thr = 8 waves (2M x 4N), tile 32M x 64N, K-step 64 (2 WMMA/step).
// A tile staged into LDS (reused by 4 N-waves) with one 16B async (or vector)
// transfer per thread; W streamed from global — all weights are L2-resident
// (~37MB bf16 << 192MB L2), so these GEMMs run at WMMA rate.
// No bounds checks anywhere; EXEC is all-ones at every WMMA.
// ---------------------------------------------------------------------------
__global__ __launch_bounds__(256)
void gemm_fast_k(const __bf16* __restrict__ A, int lda,
                 const __bf16* __restrict__ Wt, int ldw,
                 const float* __restrict__ bias,
                 float* __restrict__ Cf, __bf16* __restrict__ Cb, int ldc,
                 int K, int accum, int epi)
{
    __shared__ __bf16 As[32 * 72];   // 32 rows x 64 K, stride 72 elem (144B; 16B-aligned chunks)

    const int tid  = threadIdx.x;
    const int lane = tid & 31;
    const int wave = tid >> 5;
    const int wm   = wave & 1;
    const int wn   = wave >> 1;

    const int mBase = blockIdx.y * 32;
    const int n0    = blockIdx.x * 64 + wn * 16;
    const int nn    = n0 + (lane & 15);
    const int kbA   = (lane & 16) ? 16 : 0;   // A layout: contiguous K half (matches B)
    const int kbB   = (lane & 16) ? 16 : 0;   // B layout: contiguous K half (matches A)

    const int sr  = tid >> 3;           // staging row 0..31
    const int sc8 = (tid & 7) * 8;      // staging col 0,8,...,56

    const __bf16* aStage = A + (size_t)(mBase + sr) * lda + sc8;
    const __bf16* wRow   = Wt + (size_t)nn * ldw + kbB;
    const __bf16* arow   = &As[(wm * 16 + (lane & 15)) * 72];
    __bf16*       sdst   = &As[sr * 72 + sc8];

#if HAVE_ASYNC_LDS
    const unsigned ldsAddr =
        (unsigned)(uintptr_t)(__attribute__((address_space(3))) void*)sdst;
#endif

    v8f acc;
    #pragma unroll
    for (int i = 0; i < 8; ++i) acc[i] = 0.0f;

    for (int k0 = 0; k0 < K; k0 += 64) {
#if HAVE_ASYNC_LDS
        // 16B per lane, global -> LDS, tracked by ASYNCcnt.
        asm volatile("global_load_async_to_lds_b128 %0, %1, off"
                     :
                     : "v"(ldsAddr),
                       "v"((unsigned long long)(uintptr_t)(aStage + k0))
                     : "memory");
        asm volatile("s_wait_asynccnt 0" ::: "memory");
#else
        *(bf16x8*)sdst = *(const bf16x8*)(aStage + k0);
        if (k0 + 64 < K) __builtin_prefetch(aStage + k0 + 64, 0, 3);
#endif
        __syncthreads();

        #pragma unroll
        for (int kk = 0; kk < 2; ++kk) {
            const int ko = kk * 32;
            bf16x16 afrag;
            {
                bf16x8 a0 = *(const bf16x8*)(arow + ko + kbA);
                bf16x8 a1 = *(const bf16x8*)(arow + ko + 8 + kbA);
                #pragma unroll
                for (int i = 0; i < 8; ++i) { afrag[i] = a0[i]; afrag[8 + i] = a1[i]; }
            }
            bf16x16 bfrag;
            {
                const __bf16* wp = wRow + (k0 + ko);
                bf16x8 b0 = *(const bf16x8*)(wp);
                bf16x8 b1 = *(const bf16x8*)(wp + 8);
                #pragma unroll
                for (int i = 0; i < 8; ++i) { bfrag[i] = b0[i]; bfrag[8 + i] = b1[i]; }
            }
            acc = __builtin_amdgcn_wmma_f32_16x16x32_bf16(false, afrag, false, bfrag,
                                                          (short)0, acc, false, false);
        }
        __syncthreads();
    }

    // D layout: VGPR i -> M row (+8 for lanes>=16), lane&15 -> N col
    const int rb = mBase + wm * 16 + ((lane & 16) ? 8 : 0);
    const float bv = bias ? bias[nn] : 0.0f;
    #pragma unroll
    for (int i = 0; i < 8; ++i) {
        const size_t o = (size_t)(rb + i) * ldc + nn;
        float v = acc[i] + bv;
        if (accum) v += Cf[o];
        if (epi == 1)      v = dev_silu(v);
        else if (epi == 2) v = dev_softplus(v);
        Cf[o] = v;
        if (Cb) Cb[o] = (__bf16)v;
    }
}

// ---------------------------------------------------------------------------
// EDGE WMMA GEMM: handles arbitrary M/N/K/lda/ldw (clamped branchless loads).
// Only used for the small GEMMs: x_proj (N=44/56/68), dt (K=dr), patch (K=48),
// head (M=16). K-step 32, tile 32M x 64N.
// ---------------------------------------------------------------------------
__global__ __launch_bounds__(256)
void gemm_edge_k(const __bf16* __restrict__ A, int lda,
                 const __bf16* __restrict__ Wt, int ldw,
                 const float* __restrict__ bias,
                 float* __restrict__ Cf, __bf16* __restrict__ Cb, int ldc,
                 int M, int N, int K, int accum, int epi)
{
    __shared__ __bf16 As[32 * 40];   // 32 x 32, stride 40 elem = 80B (16B-aligned chunks)

    const int tid  = threadIdx.x;
    const int lane = tid & 31;
    const int wave = tid >> 5;
    const int wm   = wave & 1;
    const int wn   = wave >> 1;

    const int mBase = blockIdx.y * 32;
    const int n0    = blockIdx.x * 64 + wn * 16;
    const int nn    = n0 + (lane & 15);
    const int kbA   = (lane & 16) ? 16 : 0;
    const int kbB   = (lane & 16) ? 16 : 0;

    const int sr  = tid >> 3;          // 0..31
    const int sc4 = (tid & 7) * 4;     // 0,4,...,28

    const bool aVec  = ((lda & 3) == 0);
    const bool bVec  = ((ldw & 7) == 0);
    const bool mFull = (mBase + 32 <= M);       // block-uniform
    const bool nFullWave = (n0 + 16 <= N);      // wave-uniform

    const int nClamp = (nn < N) ? nn : 0;
    const __bf16* wRow = Wt + (size_t)nClamp * ldw;
    const __bf16* arow = &As[(wm * 16 + (lane & 15)) * 40];

    v8f acc;
    #pragma unroll
    for (int i = 0; i < 8; ++i) acc[i] = 0.0f;

    for (int k0 = 0; k0 < K; k0 += 32) {
        const bool kFull = (k0 + 32 <= K);      // block-uniform

        if (mFull && kFull && aVec) {
            const __bf16* ap = A + (size_t)(mBase + sr) * lda + (k0 + sc4);
            *(bf16x4*)&As[sr * 40 + sc4] = *(const bf16x4*)ap;
        } else {
            const int gm  = mBase + sr;
            const int gmc = (gm < M) ? gm : 0;
            const __bf16* ap = A + (size_t)gmc * lda;
            #pragma unroll
            for (int j = 0; j < 4; ++j) {
                const int k  = k0 + sc4 + j;
                const int kc = (k < K) ? k : 0;
                const __bf16 v = ap[kc];                       // clamped, always legal
                As[sr * 40 + sc4 + j] = (gm < M && k < K) ? v : (__bf16)0.0f;
            }
        }
        __syncthreads();

        bf16x16 afrag;
        {
            bf16x8 a0 = *(const bf16x8*)(arow + kbA);
            bf16x8 a1 = *(const bf16x8*)(arow + 8 + kbA);
            #pragma unroll
            for (int i = 0; i < 8; ++i) { afrag[i] = a0[i]; afrag[8 + i] = a1[i]; }
        }

        bf16x16 bfrag;
        if (nFullWave && kFull && bVec) {
            const __bf16* wp = wRow + (k0 + kbB);
            bf16x8 b0 = *(const bf16x8*)(wp);
            bf16x8 b1 = *(const bf16x8*)(wp + 8);
            #pragma unroll
            for (int i = 0; i < 8; ++i) { bfrag[i] = b0[i]; bfrag[8 + i] = b1[i]; }
        } else {
            #pragma unroll
            for (int i = 0; i < 16; ++i) {
                const int k  = k0 + kbB + i;
                const int kc = (k < K) ? k : 0;
                const __bf16 v = wRow[kc];                     // clamped, always legal
                bfrag[i] = (nn < N && k < K) ? v : (__bf16)0.0f;
            }
        }

        // EXEC is all-ones here (no early returns in this kernel).
        acc = __builtin_amdgcn_wmma_f32_16x16x32_bf16(false, afrag, false, bfrag,
                                                      (short)0, acc, false, false);
        __syncthreads();
    }

    const int rb = mBase + wm * 16 + ((lane & 16) ? 8 : 0);
    if (nn < N) {
        const float bv = bias ? bias[nn] : 0.0f;
        #pragma unroll
        for (int i = 0; i < 8; ++i) {
            const int r = rb + i;
            if (r < M) {
                const size_t o = (size_t)r * ldc + nn;
                float v = acc[i] + bv;
                if (accum) v += Cf[o];
                if (epi == 1)      v = dev_silu(v);
                else if (epi == 2) v = dev_softplus(v);
                Cf[o] = v;
                if (Cb) Cb[o] = (__bf16)v;
            }
        }
    }
}

// ---------------------------------------------------------------------------
// LayerNorm over rows [Nrows, C] — one wave32 per row, shuffle-tree reduction
// ---------------------------------------------------------------------------
__global__ __launch_bounds__(256)
void layernorm_k(const float* __restrict__ X, const float* __restrict__ g,
                 const float* __restrict__ b, float* __restrict__ Yf,
                 __bf16* __restrict__ Yb, int Nrows, int C)
{
    const int row  = blockIdx.x * (blockDim.x >> 5) + (threadIdx.x >> 5);
    const int lane = threadIdx.x & 31;
    if (row >= Nrows) return;   // uniform per-wave
    const float* x = X + (size_t)row * C;
    float s = 0.0f, s2 = 0.0f;
    for (int c = lane; c < C; c += 32) { const float v = x[c]; s += v; s2 += v * v; }
    #pragma unroll
    for (int o = 16; o > 0; o >>= 1) { s += __shfl_xor(s, o, 32); s2 += __shfl_xor(s2, o, 32); }
    const float m   = s / (float)C;
    const float var = s2 / (float)C - m * m;
    const float inv = rsqrtf(var + 1e-5f);
    for (int c = lane; c < C; c += 32) {
        const float v = (x[c] - m) * inv * g[c] + b[c];
        if (Yf) Yf[(size_t)row * C + c] = v;
        if (Yb) Yb[(size_t)row * C + c] = (__bf16)v;
    }
}

// ---------------------------------------------------------------------------
// Depthwise causal conv1d (k=4) + bias + SiLU. xs is cols [0,di) of xz [M,2di]
// ---------------------------------------------------------------------------
__global__ __launch_bounds__(256)
void dwconv_silu_k(const float* __restrict__ xz, int ld2,
                   const float* __restrict__ cw, const float* __restrict__ cb,
                   float* __restrict__ of, __bf16* __restrict__ ob,
                   int total, int L, int di)
{
    const int idx = blockIdx.x * blockDim.x + threadIdx.x;
    if (idx >= total) return;
    const int m = idx / di, e = idx - m * di;
    const int l = m % L;
    float a = cb[e];
    #pragma unroll
    for (int j = 0; j < 4; ++j) {
        const int lj = l - 3 + j;
        if (lj >= 0) a += cw[e * 4 + j] * xz[(size_t)(m - 3 + j) * ld2 + e];
    }
    const float v = dev_silu(a);
    of[(size_t)m * di + e] = v;
    ob[(size_t)m * di + e] = (__bf16)v;
}

// ---------------------------------------------------------------------------
// Selective scan + D skip + z-gate, fused. One thread per (batch, channel);
// 16-wide SSM state lives in registers, L sequential steps.
// ---------------------------------------------------------------------------
__global__ __launch_bounds__(256)
void scan_gate_k(const float* __restrict__ dtb, const float* __restrict__ xs,
                 const float* __restrict__ xdbl, int ldx, int dr,
                 const float* __restrict__ xz, int ld2,
                 const float* __restrict__ A_log, const float* __restrict__ Dp,
                 __bf16* __restrict__ yb, int B, int L, int di)
{
    const int idx = blockIdx.x * blockDim.x + threadIdx.x;
    if (idx >= B * di) return;
    const int b = idx / di, e = idx - b * di;
    float Av[16], h[16];
    #pragma unroll
    for (int s = 0; s < 16; ++s) { Av[s] = -__expf(A_log[e * 16 + s]); h[s] = 0.0f; }
    const float Dc = Dp[e];
    for (int l = 0; l < L; ++l) {
        const size_t m = (size_t)b * L + l;
        const float dtv = dtb[m * di + e];
        const float xv  = xs[m * di + e];
        const float* BC = xdbl + m * ldx + dr;   // Bs at [dr,dr+16), Cs at [dr+16,dr+32)
        float y = 0.0f;
        #pragma unroll
        for (int s = 0; s < 16; ++s) {
            h[s] = __expf(dtv * Av[s]) * h[s] + dtv * BC[s] * xv;
            y += h[s] * BC[16 + s];
        }
        y += Dc * xv;
        const float zv = xz[m * ld2 + di + e];
        y *= dev_silu(zv);
        yb[m * di + e] = (__bf16)y;
    }
}

// ---------------------------------------------------------------------------
// im2col for patch embed: x [16,3,128,128] -> Q [16*32*32, 48], col=c*16+i*4+j
// ---------------------------------------------------------------------------
__global__ __launch_bounds__(256)
void im2col_patch_k(const float* __restrict__ x, __bf16* __restrict__ Q, int total)
{
    const int idx = blockIdx.x * blockDim.x + threadIdx.x;
    if (idx >= total) return;
    const int col = idx % 48;
    const int row = idx / 48;
    const int w2 = row & 31, h2 = (row >> 5) & 31, b = row >> 10;
    const int c = col >> 4, i = (col >> 2) & 3, j = col & 3;
    Q[idx] = (__bf16)x[(((size_t)b * 3 + c) * 128 + (h2 * 4 + i)) * 128 + (w2 * 4 + j)];
}

// ---------------------------------------------------------------------------
// im2col for 2x2 stride-2 downsample: t [B,H*W,C] f32 -> Q [B*(H/2)*(W/2),4C]
// col = c*4 + i*2 + j  (matches weight (C2,C,2,2) flat layout)
// ---------------------------------------------------------------------------
__global__ __launch_bounds__(256)
void im2col_down_k(const float* __restrict__ t, __bf16* __restrict__ Q,
                   int total, int H, int W, int C)
{
    const int idx = blockIdx.x * blockDim.x + threadIdx.x;
    if (idx >= total) return;
    const int K4 = 4 * C;
    const int col = idx % K4;
    const int row = idx / K4;
    const int c = col >> 2, i = (col >> 1) & 1, j = col & 1;
    const int W2 = W >> 1, H2 = H >> 1;
    const int w2 = row % W2, h2 = (row / W2) % H2, b = row / (W2 * H2);
    const int l = (h2 * 2 + i) * W + (w2 * 2 + j);
    Q[idx] = (__bf16)t[((size_t)b * H * W + l) * C + c];
}

// ---------------------------------------------------------------------------
// Mean-pool tokens: t [B,L,C] f32 -> pooled bf16 [B,C]
// ---------------------------------------------------------------------------
__global__ __launch_bounds__(256)
void meanpool_k(const float* __restrict__ t, __bf16* __restrict__ p,
                int B, int L, int C)
{
    const int idx = blockIdx.x * blockDim.x + threadIdx.x;
    if (idx >= B * C) return;
    const int b = idx / C, c = idx - b * C;
    float s = 0.0f;
    for (int l = 0; l < L; ++l) s += t[((size_t)b * L + l) * C + c];
    p[idx] = (__bf16)(s / (float)L);
}

// ---------------------------------------------------------------------------
// f32 -> bf16 converter (weights), grid-stride
// ---------------------------------------------------------------------------
__global__ __launch_bounds__(256)
void f2b_k(const float* __restrict__ s, __bf16* __restrict__ d, size_t n)
{
    for (size_t i = (size_t)blockIdx.x * blockDim.x + threadIdx.x; i < n;
         i += (size_t)gridDim.x * blockDim.x)
        d[i] = (__bf16)s[i];
}

// ===========================================================================
// Host orchestration
// ===========================================================================
static inline int ceil_div(int a, int b) { return (a + b - 1) / b; }

// leaf offsets inside one mamba block (sorted-key pytree order)
enum { BO_ALOG = 0, BO_D, BO_CONVB, BO_CONVW, BO_DTB, BO_DTW, BO_INPROJ, BO_OUTPROJ,
       BO_XPROJ, BO_NORMB, BO_NORMG };

extern "C" void kernel_launch(void* const* d_in, const int* in_sizes, int n_in,
                              void* d_out, int out_size, void* d_ws, size_t ws_size,
                              hipStream_t stream)
{
    (void)in_sizes; (void)n_in; (void)out_size; (void)ws_size;

    static const int DIMS[3]   = {192, 384, 576};
    static const int DEPTHS[3] = {2, 4, 6};

    // leaf indices (sorted-key flatten): x, downs[0]{b,bn,g,w}, downs[1]{...},
    // head_b, head_w, patch_b, patch_bn, patch_g, patch_w, stages...
    const int I_DN_B[2] = {1, 5}, I_DN_BN[2] = {2, 6}, I_DN_G[2] = {3, 7}, I_DN_W[2] = {4, 8};
    const int I_HEADB = 9, I_HEADW = 10, I_PB = 11, I_PBN = 12, I_PG = 13, I_PW = 14;

    int blkBase[3][6], outB[3], outG[3];
    {
        int base = 15;
        for (int s = 0; s < 3; ++s) {
            for (int b = 0; b < DEPTHS[s]; ++b) { blkBase[s][b] = base; base += 11; }
            outB[s] = base++; outG[s] = base++;
        }
    }

    auto F = [&](int i) -> const float* { return (const float*)d_in[i]; };

    // bump allocator over workspace
    char* wsp = (char*)d_ws;
    size_t off = 0;
    auto take = [&](size_t bytes) -> void* {
        off = (off + 255) & ~(size_t)255;
        void* p = wsp + off;
        off += bytes;
        return p;
    };

    auto cvt = [&](int idx, size_t n) -> __bf16* {
        __bf16* dst = (__bf16*)take(n * sizeof(__bf16));
        int blocks = (int)((n + 255) / 256); if (blocks > 16384) blocks = 16384;
        f2b_k<<<blocks, 256, 0, stream>>>(F(idx), dst, n);
        return dst;
    };

    // dispatch: fast kernel when the tile/alignment preconditions hold
    auto gemm = [&](const __bf16* A, int lda, const __bf16* W, int ldw,
                    const float* bias, float* Cf, __bf16* Cb, int ldc,
                    int M, int N, int K, int accum, int epi) {
        dim3 g((unsigned)ceil_div(N, 64), (unsigned)ceil_div(M, 32));
        const bool fast = (M % 32 == 0) && (N % 64 == 0) && (K % 64 == 0) &&
                          ((lda & 7) == 0) && ((ldw & 7) == 0);
        if (fast)
            gemm_fast_k<<<g, 256, 0, stream>>>(A, lda, W, ldw, bias, Cf, Cb, ldc,
                                               K, accum, epi);
        else
            gemm_edge_k<<<g, 256, 0, stream>>>(A, lda, W, ldw, bias, Cf, Cb, ldc,
                                               M, N, K, accum, epi);
    };
    auto ln = [&](const float* X, const float* g, const float* b,
                  float* Yf, __bf16* Yb, int Nr, int C) {
        layernorm_k<<<ceil_div(Nr, 8), 256, 0, stream>>>(X, g, b, Yf, Yb, Nr, C);
    };

    // ---- persistent activation scratch (sized for stage-0 maxima) ----
    const int MMAX = 16 * 1024;
    float*  t_f   = (float*)take((size_t)MMAX * 192 * 4);   // tokens f32, max M*d
    __bf16* tn_b  = (__bf16*)take((size_t)MMAX * 192 * 2);  // LN(t) bf16
    float*  xz_f  = (float*)take((size_t)MMAX * 768 * 4);   // in_proj out, max M*2di
    float*  xs_f  = (float*)take((size_t)MMAX * 384 * 4);   // conv+silu f32
    __bf16* xs_b  = (__bf16*)take((size_t)MMAX * 384 * 2);
    float*  xd_f  = (float*)take((size_t)MMAX * 44 * 4);    // x_dbl f32 (max M*(dr+32))
    __bf16* xd_b  = (__bf16*)take((size_t)MMAX * 44 * 2);
    float*  dt_f  = (float*)take((size_t)MMAX * 384 * 4);   // softplus(dt) f32
    __bf16* y_b   = (__bf16*)take((size_t)MMAX * 384 * 2);  // gated scan out bf16
    __bf16* q_b   = (__bf16*)take((size_t)4096 * 768 * 2);  // im2col staging (max)
    __bf16* pool_b= (__bf16*)take((size_t)16 * 576 * 2);

    // ---- patch embed: im2col + WMMA GEMM + LN ----
    {
        const int M = 16 * 32 * 32, K = 48, N = 192;
        const int total = M * K;
        im2col_patch_k<<<ceil_div(total, 256), 256, 0, stream>>>(F(0), q_b, total);
        __bf16* wp = cvt(I_PW, (size_t)N * K);
        gemm(q_b, K, wp, K, F(I_PB), t_f, nullptr, N, M, N, K, 0, 0);
        ln(t_f, F(I_PG), F(I_PBN), t_f, nullptr, M, N);
    }

    int H = 32, W = 32;
    for (int s = 0; s < 3; ++s) {
        const int d  = DIMS[s];
        const int di = 2 * d;
        const int dr = (d + 15) / 16;
        const int L  = H * W;
        const int M  = 16 * L;
        const int ldx = dr + 32;

        for (int bk = 0; bk < DEPTHS[s]; ++bk) {
            const int base = blkBase[s][bk];
            __bf16* w_in  = cvt(base + BO_INPROJ,  (size_t)2 * di * d);
            __bf16* w_out = cvt(base + BO_OUTPROJ, (size_t)d * di);
            __bf16* w_xp  = cvt(base + BO_XPROJ,   (size_t)ldx * di);
            __bf16* w_dt  = cvt(base + BO_DTW,     (size_t)di * dr);

            // pre-norm -> bf16 activations
            ln(t_f, F(base + BO_NORMG), F(base + BO_NORMB), nullptr, tn_b, M, d);
            // in_proj: [M,d] x [d,2di]  (fast kernel)
            gemm(tn_b, d, w_in, d, nullptr, xz_f, nullptr, 2 * di, M, 2 * di, d, 0, 0);
            // depthwise causal conv + silu
            {
                const int total = M * di;
                dwconv_silu_k<<<ceil_div(total, 256), 256, 0, stream>>>(
                    xz_f, 2 * di, F(base + BO_CONVW), F(base + BO_CONVB),
                    xs_f, xs_b, total, L, di);
            }
            // x_proj: [M,di] x [di, dr+32]  (edge kernel: N=44/56/68)
            gemm(xs_b, di, w_xp, di, nullptr, xd_f, xd_b, ldx, M, ldx, di, 0, 0);
            // dt: [M,dr] x [dr,di] + dt_b, softplus  (edge kernel: K=dr)
            gemm(xd_b, ldx, w_dt, dr, F(base + BO_DTB), dt_f, nullptr, di, M, di, dr, 0, 2);
            // selective scan + D skip + z gate
            {
                const int total = 16 * di;
                scan_gate_k<<<ceil_div(total, 256), 256, 0, stream>>>(
                    dt_f, xs_f, xd_f, ldx, dr, xz_f, 2 * di,
                    F(base + BO_ALOG), F(base + BO_D), y_b, 16, L, di);
            }
            // out_proj with fused residual: t += y @ out_proj^T  (fast kernel)
            gemm(y_b, di, w_out, di, nullptr, t_f, nullptr, d, M, d, di, 1, 0);
        }

        // stage-end LN (in place)
        ln(t_f, F(outG[s]), F(outB[s]), t_f, nullptr, M, d);

        if (s < 2) {
            const int C2 = DIMS[s + 1];
            const int M2 = 16 * (H / 2) * (W / 2);
            const int K2 = 4 * d;
            const int total = M2 * K2;
            im2col_down_k<<<ceil_div(total, 256), 256, 0, stream>>>(t_f, q_b, total, H, W, d);
            __bf16* w_dn = cvt(I_DN_W[s], (size_t)C2 * K2);
            gemm(q_b, K2, w_dn, K2, F(I_DN_B[s]), t_f, nullptr, C2, M2, C2, K2, 0, 0);
            ln(t_f, F(I_DN_G[s]), F(I_DN_BN[s]), t_f, nullptr, M2, C2);
            H >>= 1; W >>= 1;
        }
    }

    // ---- head: mean pool + GEMM [16,576]x[576,1000]  (edge kernel: M=16) ----
    {
        const int B = 16, L = H * W /* 64 */, C = 576, N = 1000;
        meanpool_k<<<ceil_div(B * C, 256), 256, 0, stream>>>(t_f, pool_b, B, L, C);
        __bf16* w_h = cvt(I_HEADW, (size_t)N * C);
        gemm(pool_b, C, w_h, C, F(I_HEADB), (float*)d_out, nullptr, N, B, N, C, 0, 0);
    }
}
